// Block_12781822673211
// MI455X (gfx1250) — matrix operand order repaired and checked by reading, not compiled, not run on hardware
//
#include <hip/hip_runtime.h>

// ---------------------------------------------------------------------------
// MI455X (gfx1250) transformer block, bf16 WMMA (v_wmma_f32_16x16x32_bf16)
// Round 3: GEMM k-loop unrolled 2x with ping-pong fragment buffers (no
// register rotation copies, fewer hazard NOPs); prefetch kept near (loc=3).
// ---------------------------------------------------------------------------

typedef __attribute__((ext_vector_type(16))) __bf16 v16bf;
typedef __attribute__((ext_vector_type(8)))  float  v8f;
typedef __attribute__((ext_vector_type(4)))  unsigned int v4u;

#define DEV static __device__ __forceinline__

constexpr int Cc = 384;     // n_embed
constexpr int Hh = 6;       // heads
constexpr int Dd = 64;      // head dim
constexpr int Bbatch = 4;
constexpr int Tt = 2048;
constexpr int BT = Bbatch * Tt;  // 8192 rows
constexpr int FF = 4 * Cc;       // 1536

DEV int lane_id() { return (int)(threadIdx.x & 31u); }

DEV v8f wmma_bf16(v16bf a, v16bf b, v8f c) {
  // D(16x16,f32) = A(16x32,bf16) x B(32x16,bf16) + C
  return __builtin_amdgcn_wmma_f32_16x16x32_bf16(
      /*neg_a=*/false, a, /*neg_b=*/false, b,
      /*c_mod=*/(short)0, c, /*reuse_a=*/false, /*reuse_b=*/false);
}

union AF { v16bf v; unsigned int u[8]; unsigned short s[16]; };
union BF { v16bf v; v4u q[2];          unsigned short s[16]; };

// A fragment (16x32 bf16, MxK) from row-major [rows x lda] bf16 matrix.
// ISA layout: lane L holds row (L&15); lanes 0-15 K={0..7,16..23},
// lanes 16-31 K={8..15,24..31}; 2 elems per VGPR -> dword loads.
DEV v16bf load_a_frag(const __bf16* base, int lda, int m0, int k0) {
  const int l = lane_id(), hl = l >> 4, row = m0 + (l & 15);
  const __bf16* p = base + (size_t)row * lda + k0 + hl * 8;
  AF f;
  const unsigned int* p32 = (const unsigned int*)p;        // K = hl*8 + {0..7}
  f.u[0] = p32[0]; f.u[1] = p32[1]; f.u[2] = p32[2]; f.u[3] = p32[3];
  const unsigned int* q32 = (const unsigned int*)(p + 16); // K = 16 + hl*8 + {0..7}
  f.u[4] = q32[0]; f.u[5] = q32[1]; f.u[6] = q32[2]; f.u[7] = q32[3];
  return f.v;
}

// B fragment (32x16 bf16, KxN) from a matrix stored [N x ldw] (i.e. W^T,
// row = output col, contiguous in K). Lane L holds col (L&15);
// lanes 0-15 K={0..7,16..23}, lanes 16-31 K={8..15,24..31} -> two b128 loads.
DEV v16bf load_b_frag(const __bf16* wt, int ldw, int n0, int k0) {
  const int l = lane_id(), hl = l >> 4, col = n0 + (l & 15);
  const __bf16* p = wt + (size_t)col * ldw + k0 + hl * 8;
  BF f;
  const v4u* p128 = (const v4u*)p;
  f.q[0] = p128[0]; f.q[1] = p128[2];
  return f.v;
}

// P (16x32 probs, row-major bf16 in LDS, ld=32) as a full A fragment.
DEV v16bf load_ap_frag32(const __bf16* lds) {
  const int l = lane_id(), hl = l >> 4, m = l & 15;
  BF f;
  f.q[0] = *(const v4u*)(lds + m * 32 + hl * 8);        // K = hl*8  + {0..7}
  f.q[1] = *(const v4u*)(lds + m * 32 + 16 + hl * 8);   // K = 16+hl*8 + {0..7}
  return f.v;
}

// V tile (32 rows s0.., 16 cols n0..) as a full B fragment (K = s index).
// V row-major [*, Dd]; per-lane strided (stride Dd), no divergence.
DEV v16bf load_bv_frag32(const __bf16* vbase, int s0, int n0) {
  const int l = lane_id(), hl = l >> 4, col = n0 + (l & 15);
  const unsigned short* vs = (const unsigned short*)vbase;
  BF f;
#pragma unroll
  for (int e = 0; e < 16; ++e)
    f.s[e] = vs[(size_t)(s0 + (e < 8 ? hl * 8 + e : 16 + hl * 8 + (e - 8))) * Dd + col];
  return f.v;
}

// ---------------------------------------------------------------------------
// Weight convert + transpose: in fp32 [K x N] -> out bf16 [N x K] (batched).
// ---------------------------------------------------------------------------
__global__ void transpose_bf16_kernel(const float* __restrict__ in,
                                      __bf16* __restrict__ out, int K, int N) {
  const size_t mat = (size_t)K * N;
  const float* ib = in + (size_t)blockIdx.y * mat;
  __bf16* ob = out + (size_t)blockIdx.y * mat;
  for (int i = blockIdx.x * blockDim.x + threadIdx.x; i < K * N;
       i += gridDim.x * blockDim.x) {
    const int k = i / N, n = i % N;
    ob[(size_t)n * K + k] = (__bf16)ib[i];
  }
}

// ---------------------------------------------------------------------------
// LayerNorm: one wave per row of length Cc; fp32 in -> bf16 out.
// ---------------------------------------------------------------------------
__global__ void __launch_bounds__(32)
ln_kernel(const float* __restrict__ x, const float* __restrict__ g,
          const float* __restrict__ bta, __bf16* __restrict__ out) {
  const int row = blockIdx.x, l = lane_id();
  const float* xr = x + (size_t)row * Cc;
  float s = 0.f, s2 = 0.f;
  for (int c = l; c < Cc; c += 32) { const float v = xr[c]; s += v; s2 += v * v; }
  for (int m = 1; m < 32; m <<= 1) { s += __shfl_xor(s, m, 32); s2 += __shfl_xor(s2, m, 32); }
  const float mu = s / Cc;
  const float rstd = rsqrtf(s2 / Cc - mu * mu + 1e-5f);
  for (int c = l; c < Cc; c += 32)
    out[(size_t)row * Cc + c] = (__bf16)((xr[c] - mu) * rstd * g[c] + bta[c]);
}

// ---------------------------------------------------------------------------
// bf16 WMMA GEMM: out[M x N] = A[M x K] * W[K x N] (+bias)(+resid)(relu)
// W pre-transposed bf16 [N x K].  One wave -> 32x64 tile.
// K must be a multiple of 64 (here 384 or 1536).  Ping-pong double buffering:
// even k-tiles use stage-0 fragment registers, odd k-tiles stage-1; loads for
// one stage issue while the other stage's 8 WMMAs execute, no rotation moves.
// ---------------------------------------------------------------------------
__global__ void __launch_bounds__(32)
gemm_bf16_kernel(const __bf16* __restrict__ A, const __bf16* __restrict__ Wt,
                 int K, int lda,
                 long long aBatch, long long bBatch, long long oBatch,
                 const float* __restrict__ bias,
                 const float* __restrict__ resid, int ldr,
                 float* __restrict__ outF, __bf16* __restrict__ outB,
                 int ldo, int relu) {
  const int m0 = blockIdx.x * 32;
  const int n0 = blockIdx.y * 64;
  const int bz = blockIdx.z;
  const __bf16* Ab = A + (size_t)bz * aBatch;
  const __bf16* Wb = Wt + (size_t)bz * bBatch;

  v8f acc[2][4] = {};
  v16bf A0[2], B0[4], A1[2], B1[4];   // ping-pong fragment stages

  auto load_stage = [&](v16bf* Af, v16bf* Bf, int k0) {
    Af[0] = load_a_frag(Ab, lda, m0, k0);
    Af[1] = load_a_frag(Ab, lda, m0 + 16, k0);
#pragma unroll
    for (int j = 0; j < 4; ++j) Bf[j] = load_b_frag(Wb, K, n0 + j * 16, k0);
  };
  auto mma_stage = [&](const v16bf* Af, const v16bf* Bf) {
#pragma unroll
    for (int j = 0; j < 4; ++j) {
      acc[0][j] = wmma_bf16(Af[0], Bf[j], acc[0][j]);
      acc[1][j] = wmma_bf16(Af[1], Bf[j], acc[1][j]);
    }
  };

  load_stage(A0, B0, 0);
  int k0 = 0;
  for (; k0 + 64 < K; k0 += 64) {
    load_stage(A1, B1, k0 + 32);      // odd tile loads overlap even WMMAs
    __builtin_prefetch(Ab + (size_t)(m0 + (lane_id() & 15)) * lda + k0 + 64,
                       0, 3);         // keep-near prefetch of next even tile
    mma_stage(A0, B0);
    load_stage(A0, B0, k0 + 64);      // even tile loads overlap odd WMMAs
    mma_stage(A1, B1);
  }
  // tail: k0 == K-64; stage0 (k0) already loaded
  load_stage(A1, B1, k0 + 32);
  mma_stage(A0, B0);
  mma_stage(A1, B1);

  const int l = lane_id(), hl = l >> 4, nc = l & 15;
#pragma unroll
  for (int hm = 0; hm < 2; ++hm) {
#pragma unroll
    for (int j = 0; j < 4; ++j) {
      const int col = n0 + j * 16 + nc;
      const float bv = bias ? bias[col] : 0.0f;
#pragma unroll
      for (int r = 0; r < 8; ++r) {
        const int row = m0 + hm * 16 + r + 8 * hl;  // C/D: VGPR r -> M = r+8*half
        float v = acc[hm][j][r] + bv;
        if (resid) v += resid[(size_t)row * ldr + col];
        if (relu)  v = fmaxf(v, 0.0f);
        const size_t oi = (size_t)bz * oBatch + (size_t)row * ldo + col;
        if (outF) outF[oi] = v;
        else      outB[oi] = (__bf16)v;
      }
    }
  }
}

// ---------------------------------------------------------------------------
// Flash-style causal attention.  q = v = qv (source quirk), SCALE = 0.25.
// One wave per (b, h, 16-row query tile).  32 keys per iteration: two S
// subtiles, online softmax, P staged as 16x32 bf16 in LDS -> full-K PV WMMA.
// ---------------------------------------------------------------------------
__global__ void __launch_bounds__(32)
attention_kernel(const __bf16* __restrict__ kbf, const __bf16* __restrict__ qvbf,
                 __bf16* __restrict__ attn) {
  __shared__ __bf16 ldsP[16 * 32];  // 1 KB staging: C-layout -> A-layout for P
  const int t0 = blockIdx.x * 16;
  const int b  = blockIdx.y / Hh;
  const int h  = blockIdx.y % Hh;
  const __bf16* k_h  = kbf  + ((size_t)h * BT + (size_t)b * Tt) * Dd;
  const __bf16* qv_h = qvbf + ((size_t)h * BT + (size_t)b * Tt) * Dd;

  const int l = lane_id(), hl = l >> 4, nc = l & 15;

  // Q A-fragments (reused across all key tiles): rows t0.., K = head dim 0..63
  const v16bf aQ0 = load_a_frag(qv_h, Dd, t0, 0);
  const v16bf aQ1 = load_a_frag(qv_h, Dd, t0, 32);

  float m_i[8], l_i[8];
#pragma unroll
  for (int r = 0; r < 8; ++r) { m_i[r] = -1e30f; l_i[r] = 0.0f; }
  v8f o[4] = {};

  // covers keys s0..s0+31; diagonal raggedness handled by the mask.
  for (int s0 = 0; s0 <= t0; s0 += 32) {
    // S subtiles: cols s0..s0+15 and s0+16..s0+31
    v8f sa = {}, sb = {};
    sa = wmma_bf16(aQ0, load_b_frag(k_h, Dd, s0, 0),       sa);
    sa = wmma_bf16(aQ1, load_b_frag(k_h, Dd, s0, 32),      sa);
    sb = wmma_bf16(aQ0, load_b_frag(k_h, Dd, s0 + 16, 0),  sb);
    sb = wmma_bf16(aQ1, load_b_frag(k_h, Dd, s0 + 16, 32), sb);

    float pa[8], pb[8];
#pragma unroll
    for (int r = 0; r < 8; ++r) {
      const int t = t0 + r + 8 * hl;          // query row of this C element
      pa[r] = (s0 + nc      > t) ? -1e30f : sa[r] * 0.25f;  // SCALE = 16^-0.5
      pb[r] = (s0 + 16 + nc > t) ? -1e30f : sb[r] * 0.25f;
    }
    // row max over the 16 lanes holding each row (XOR masks stay in-half)
    float rmx[8];
#pragma unroll
    for (int r = 0; r < 8; ++r) rmx[r] = fmaxf(pa[r], pb[r]);
    for (int m = 1; m < 16; m <<= 1)
#pragma unroll
      for (int r = 0; r < 8; ++r) rmx[r] = fmaxf(rmx[r], __shfl_xor(rmx[r], m, 32));

    float alpha[8];
#pragma unroll
    for (int r = 0; r < 8; ++r) {
      const float mn = fmaxf(m_i[r], rmx[r]);
      alpha[r] = __expf(m_i[r] - mn);
      m_i[r] = mn;
      pa[r] = __expf(pa[r] - mn);             // masked -> 0
      pb[r] = __expf(pb[r] - mn);
    }
    float rsm[8];
#pragma unroll
    for (int r = 0; r < 8; ++r) rsm[r] = pa[r] + pb[r];
    for (int m = 1; m < 16; m <<= 1)
#pragma unroll
      for (int r = 0; r < 8; ++r) rsm[r] += __shfl_xor(rsm[r], m, 32);

#pragma unroll
    for (int r = 0; r < 8; ++r) l_i[r] = l_i[r] * alpha[r] + rsm[r];
#pragma unroll
    for (int dt = 0; dt < 4; ++dt)
#pragma unroll
      for (int r = 0; r < 8; ++r) o[dt][r] *= alpha[r];

    // re-layout P: C fragments -> row-major 16x32 bf16 tile in LDS -> A frag
    __syncthreads();
#pragma unroll
    for (int r = 0; r < 8; ++r) {
      ldsP[(r + 8 * hl) * 32 + nc]      = (__bf16)pa[r];
      ldsP[(r + 8 * hl) * 32 + 16 + nc] = (__bf16)pb[r];
    }
    __syncthreads();

    const v16bf aP = load_ap_frag32(ldsP);
#pragma unroll
    for (int dt = 0; dt < 4; ++dt)
      o[dt] = wmma_bf16(aP, load_bv_frag32(qv_h, s0, dt * 16), o[dt]);
  }

  // write concat-head layout: attn[b*T + t][h*64 + d], normalized by l_i
  const int bt0 = b * Tt + t0;
#pragma unroll
  for (int dt = 0; dt < 4; ++dt)
#pragma unroll
    for (int r = 0; r < 8; ++r) {
      const int row = bt0 + r + 8 * hl;
      attn[(size_t)row * Cc + h * Dd + dt * 16 + nc] = (__bf16)(o[dt][r] / l_i[r]);
    }
}

// ---------------------------------------------------------------------------
// Host-side orchestration
// ---------------------------------------------------------------------------
extern "C" void kernel_launch(void* const* d_in, const int* in_sizes, int n_in,
                              void* d_out, int out_size, void* d_ws, size_t ws_size,
                              hipStream_t stream) {
  (void)in_sizes; (void)n_in; (void)out_size; (void)ws_size;
  const float* x      = (const float*)d_in[0];
  const float* ln1_g  = (const float*)d_in[1];
  const float* ln1_b  = (const float*)d_in[2];
  const float* wk     = (const float*)d_in[3];   // (H, C, D)
  const float* wv     = (const float*)d_in[4];   // (H, C, D)
  const float* w_proj = (const float*)d_in[5];   // (C, C)
  const float* b_proj = (const float*)d_in[6];
  const float* ln2_g  = (const float*)d_in[7];
  const float* ln2_b  = (const float*)d_in[8];
  const float* w_ff1  = (const float*)d_in[9];   // (C, 4C)
  const float* b_ff1  = (const float*)d_in[10];
  const float* w_ff2  = (const float*)d_in[11];  // (4C, C)
  const float* b_ff2  = (const float*)d_in[12];
  float* out = (float*)d_out;

  // workspace carve-up (all sizes already 256B-aligned)
  char* ws = (char*)d_ws;
  auto take = [&](size_t bytes) -> char* {
    char* p = ws; ws += (bytes + 255) & ~(size_t)255; return p;
  };
  __bf16* wkT  = (__bf16*)take((size_t)Hh * Dd * Cc * 2);   // [h][d][c]
  __bf16* wvT  = (__bf16*)take((size_t)Hh * Dd * Cc * 2);
  __bf16* wpT  = (__bf16*)take((size_t)Cc * Cc * 2);        // [n][k]
  __bf16* wf1T = (__bf16*)take((size_t)Cc * FF * 2);
  __bf16* wf2T = (__bf16*)take((size_t)FF * Cc * 2);
  __bf16* h1   = (__bf16*)take((size_t)BT * Cc * 2);        // LN1 out
  __bf16* kbf  = (__bf16*)take((size_t)Hh * BT * Dd * 2);   // [h][bt][d]
  __bf16* qvbf = (__bf16*)take((size_t)Hh * BT * Dd * 2);
  __bf16* attn = (__bf16*)take((size_t)BT * Cc * 2);        // concat heads
  float*  xo   = (float*) take((size_t)BT * Cc * 4);        // x + attn@Wp + b
  __bf16* h2   = (__bf16*)take((size_t)BT * Cc * 2);        // LN2 out
  __bf16* a1   = (__bf16*)take((size_t)BT * FF * 2);        // relu(ff1)

  // 1) weights -> bf16 transposed
  transpose_bf16_kernel<<<dim3((Cc * Dd + 255) / 256, Hh), 256, 0, stream>>>(wk, wkT, Cc, Dd);
  transpose_bf16_kernel<<<dim3((Cc * Dd + 255) / 256, Hh), 256, 0, stream>>>(wv, wvT, Cc, Dd);
  transpose_bf16_kernel<<<dim3((Cc * Cc + 255) / 256, 1), 256, 0, stream>>>(w_proj, wpT, Cc, Cc);
  transpose_bf16_kernel<<<dim3((Cc * FF + 255) / 256, 1), 256, 0, stream>>>(w_ff1, wf1T, Cc, FF);
  transpose_bf16_kernel<<<dim3((FF * Cc + 255) / 256, 1), 256, 0, stream>>>(w_ff2, wf2T, FF, Cc);

  // 2) LN1
  ln_kernel<<<dim3(BT), dim3(32), 0, stream>>>(x, ln1_g, ln1_b, h1);

  // 3) k = h1 @ wk[h], qv = h1 @ wv[h]   (batched over heads)
  gemm_bf16_kernel<<<dim3(BT / 32, Dd / 64, Hh), dim3(32), 0, stream>>>(
      h1, wkT, Cc, Cc, 0LL, (long long)Dd * Cc, (long long)BT * Dd,
      nullptr, nullptr, 0, nullptr, kbf, Dd, 0);
  gemm_bf16_kernel<<<dim3(BT / 32, Dd / 64, Hh), dim3(32), 0, stream>>>(
      h1, wvT, Cc, Cc, 0LL, (long long)Dd * Cc, (long long)BT * Dd,
      nullptr, nullptr, 0, nullptr, qvbf, Dd, 0);

  // 4) causal flash attention (q = v = qv)
  attention_kernel<<<dim3(Tt / 16, Bbatch * Hh), dim3(32), 0, stream>>>(kbf, qvbf, attn);

  // 5) xo = x + attn @ w_proj + b_proj     (fp32 out)
  gemm_bf16_kernel<<<dim3(BT / 32, Cc / 64, 1), dim3(32), 0, stream>>>(
      attn, wpT, Cc, Cc, 0LL, 0LL, 0LL,
      b_proj, x, Cc, xo, nullptr, Cc, 0);

  // 6) LN2
  ln_kernel<<<dim3(BT), dim3(32), 0, stream>>>(xo, ln2_g, ln2_b, h2);

  // 7) a1 = relu(h2 @ w_ff1 + b_ff1)       (bf16 out)
  gemm_bf16_kernel<<<dim3(BT / 32, FF / 64, 1), dim3(32), 0, stream>>>(
      h2, wf1T, Cc, Cc, 0LL, 0LL, 0LL,
      b_ff1, nullptr, 0, nullptr, a1, FF, 1);

  // 8) out = xo + a1 @ w_ff2 + b_ff2       (fp32 out)
  gemm_bf16_kernel<<<dim3(BT / 32, Cc / 64, 1), dim3(32), 0, stream>>>(
      a1, wf2T, FF, FF, 0LL, 0LL, 0LL,
      b_ff2, xo, Cc, out, nullptr, Cc, 0);
}
